// HGNNEncoder_3015067042495
// MI455X (gfx1250) — matrix operand order repaired and checked by reading, not compiled, not run on hardware
//
#include <hip/hip_runtime.h>

// ---------------------------------------------------------------------------
// Types for CDNA5 WMMA (wave32): v16bf A/B fragments, v8f C/D accumulators.
// ---------------------------------------------------------------------------
typedef __attribute__((ext_vector_type(16))) __bf16 bf16x16;
typedef __attribute__((ext_vector_type(8)))  float  f32x8;
typedef __attribute__((ext_vector_type(4)))  unsigned int u32x4;
typedef __attribute__((ext_vector_type(8)))  int   i32x8;
typedef __attribute__((ext_vector_type(4)))  int   i32x4;

union FragU { bf16x16 v; uint4 q[2]; };
union BV4   { uint2 u; __bf16 b[4]; };

// A-fragment (16x32 bf16, MxK): lane m=lane&15; lanes16-31 take K+8;
// VGPR0-3 = K[kb..kb+7], VGPR4-7 = K[kb+16..kb+23]  (ISA 7.12.2)
__device__ __forceinline__ bf16x16 load_a_frag(const __bf16* p, int ld, int row0, int k0) {
    const int lane = threadIdx.x & 31;
    const __bf16* base = p + (size_t)(row0 + (lane & 15)) * ld + k0 + ((lane & 16) ? 8 : 0);
    FragU f;
    f.q[0] = *(const uint4*)(base);
    f.q[1] = *(const uint4*)(base + 16);
    return f.v;
}

// B-fragment (32x16 bf16, KxN) from transposed weights Wt[n][k] row-major:
// lane n=lane&15; lanes16-31 take K+8; VGPR0-3 = K[kb..kb+7], VGPR4-7 = K[kb+16..kb+23]
__device__ __forceinline__ bf16x16 load_b_frag(const __bf16* p, int ld, int col0, int k0) {
    const int lane = threadIdx.x & 31;
    const __bf16* base = p + (size_t)(col0 + (lane & 15)) * ld + k0 + ((lane & 16) ? 8 : 0);
    FragU f;
    f.q[0] = *(const uint4*)(base);
    f.q[1] = *(const uint4*)(base + 16);
    return f.v;
}

__device__ __forceinline__ f32x8 wmma_bf16(bf16x16 a, bf16x16 b, f32x8 c) {
    return __builtin_amdgcn_wmma_f32_16x16x32_bf16(false, a, false, b, (short)0, c, false, false);
}

// ---------------------------------------------------------------------------
// Tensor Data Mover: DMA a 2-D tile [128 rows x Kp bf16] from global into LDS.
// D# built per ISA 8.3/8.4 (group0: count/lds/global/type; group1: dims).
// ---------------------------------------------------------------------------
__device__ __forceinline__ void tdm_load_weights(const __bf16* Wt, int Kp, void* lds) {
    const uint64_t ga = (uint64_t)(uintptr_t)Wt;
    const uint32_t la = (uint32_t)(uintptr_t)lds;   // generic addr low 32 = LDS byte offset
    u32x4 g0;
    g0[0] = 1u;                                           // count=1, user descriptor
    g0[1] = la;                                           // lds_addr [63:32]
    g0[2] = (uint32_t)ga;                                 // global_addr low   [95:64]
    g0[3] = ((uint32_t)(ga >> 32) & 0x01FFFFFFu) | (2u << 30); // addr hi + type=2
    i32x8 g1;
    g1[0] = 0x00010000;                                   // data_size=1 (2 bytes)
    g1[1] = (int)((unsigned)Kp << 16);                    // tensor_dim0 low16 @ [63:48]
    g1[2] = (int)((128u << 16) | ((unsigned)Kp >> 16));   // dim0 hi16, tensor_dim1=128
    g1[3] = (int)((unsigned)Kp << 16);                    // tile_dim0 = Kp
    g1[4] = 128;                                          // tile_dim1 = 128
    g1[5] = Kp;                                           // tensor_dim0_stride low32
    g1[6] = 0;
    g1[7] = 0;
    i32x4 g2 = {0, 0, 0, 0};                              // dims 2/3 unused (tile_dim2=0)
    i32x4 g3 = {0, 0, 0, 0};
#if __clang_major__ >= 23
    i32x8 g4 = {0, 0, 0, 0, 0, 0, 0, 0};
    __builtin_amdgcn_tensor_load_to_lds(g0, g1, g2, g3, g4, 0);
#else
    __builtin_amdgcn_tensor_load_to_lds(g0, g1, g2, g3, 0);
#endif
}

// ---------------------------------------------------------------------------
// Pack kernels
// ---------------------------------------------------------------------------
// W [K x 128] f32  ->  Wt [128 x Kp] bf16 (transposed, zero padded)
__global__ __launch_bounds__(256) void pack_wT(const float* __restrict__ W, int K, int Kp,
                                               __bf16* __restrict__ out) {
    int idx = blockIdx.x * 256 + threadIdx.x;      // < 128*Kp
    int n = idx / Kp, k = idx % Kp;
    float v = (k < K) ? W[(size_t)k * 128 + n] : 0.0f;
    out[(size_t)n * Kp + k] = (__bf16)v;
}

// X [R x K] f32 -> out [R x Kp] bf16 zero padded
__global__ __launch_bounds__(256) void pack_pad_bf16(const float* __restrict__ X, int K, int Kp,
                                                     __bf16* __restrict__ out) {
    size_t idx = (size_t)blockIdx.x * 256 + threadIdx.x;   // < R*Kp
    size_t r = idx / Kp; int k = (int)(idx % Kp);
    float v = (k < K) ? X[r * K + k] : 0.0f;
    out[idx] = (__bf16)v;
}

// a_input = concat(f_atoms[A,133], amsg[A,128]) -> bf16 [A, 288] zero padded
__global__ __launch_bounds__(256) void pack_ainput(const float* __restrict__ f_atoms,
                                                   const __bf16* __restrict__ amsg,
                                                   __bf16* __restrict__ out) {
    size_t idx = (size_t)blockIdx.x * 256 + threadIdx.x;   // < A*288
    size_t a = idx / 288; int k = (int)(idx % 288);
    __bf16 v;
    if (k < 133)      v = (__bf16)f_atoms[a * 133 + k];
    else if (k < 261) v = amsg[a * 128 + (k - 133)];
    else              v = (__bf16)0.0f;
    out[idx] = v;
}

// ---------------------------------------------------------------------------
// amsg[a][h] = sum_{j<6} msg[a2b[a][j]][h]     (bf16 in, f32 acc, bf16 out)
// ---------------------------------------------------------------------------
__global__ __launch_bounds__(256) void gather_amsg(const __bf16* __restrict__ msg,
                                                   const int* __restrict__ a2b,
                                                   __bf16* __restrict__ amsg) {
    int idx = blockIdx.x * 256 + threadIdx.x;   // A*32 threads, 4 h each
    int a = idx >> 5;
    int h0 = (idx & 31) << 2;
    float s0 = 0.f, s1 = 0.f, s2 = 0.f, s3 = 0.f;
#pragma unroll
    for (int j = 0; j < 6; ++j) {
        int b = a2b[a * 6 + j];
        BV4 v; v.u = *(const uint2*)(msg + (size_t)b * 128 + h0);
        s0 += (float)v.b[0]; s1 += (float)v.b[1];
        s2 += (float)v.b[2]; s3 += (float)v.b[3];
    }
    BV4 o;
    o.b[0] = (__bf16)s0; o.b[1] = (__bf16)s1; o.b[2] = (__bf16)s2; o.b[3] = (__bf16)s3;
    *(uint2*)(amsg + (size_t)a * 128 + h0) = o.u;
}

// bmsg[b][h] = amsg[b2a[b]][h] - msg[b2revb[b]][h]
__global__ __launch_bounds__(256) void bond_msg(const __bf16* __restrict__ amsg,
                                                const __bf16* __restrict__ msg,
                                                const int* __restrict__ b2a,
                                                const int* __restrict__ b2revb,
                                                __bf16* __restrict__ out) {
    int idx = blockIdx.x * 256 + threadIdx.x;   // B*32 threads
    int b = idx >> 5;
    int h0 = (idx & 31) << 2;
    int a  = b2a[b];
    int rb = b2revb[b];
    BV4 x; x.u = *(const uint2*)(amsg + (size_t)a  * 128 + h0);
    BV4 y; y.u = *(const uint2*)(msg  + (size_t)rb * 128 + h0);
    BV4 o;
#pragma unroll
    for (int i = 0; i < 4; ++i) o.b[i] = (__bf16)((float)x.b[i] - (float)y.b[i]);
    *(uint2*)(out + (size_t)b * 128 + h0) = o.u;
}

// ---------------------------------------------------------------------------
// GEMM: out[M x 128] = epilogue( A[M x KP(bf16)] @ Wt^T ), Wt is [128 x KP].
// Weights DMA'd to LDS by TDM; B-fragments come from LDS (ds_load_b128).
// block = 256 threads (8 waves); block tile 128x128;
// wave tile 32 rows x 64 cols (2 m-tiles x 4 n-tiles) -> 8 WMMA / k-step.
// MODE 0: out1 = v (raw), out0 = relu(v)          (bonds -> inputs, message0)
// MODE 1: out0 = relu(v + addend)                 (message update)
// MODE 2: out0 = relu(v + bias[col])              (atom hiddens)
// ---------------------------------------------------------------------------
template <int KSTEPS, int MODE, int KP>
__global__ __launch_bounds__(256) void gemm_bf16(const __bf16* __restrict__ A,
                                                 const __bf16* __restrict__ Wt,
                                                 const __bf16* __restrict__ addend,
                                                 const float* __restrict__ bias,
                                                 __bf16* __restrict__ out0,
                                                 __bf16* __restrict__ out1) {
    extern __shared__ char smem_raw[];
    __bf16* wlds = (__bf16*)smem_raw;             // [128 x KP]

    const int wave  = threadIdx.x >> 5;
    const int lane  = threadIdx.x & 31;
    const int rows0 = blockIdx.x * 128 + (wave & 3) * 32;
    const int nb    = (wave >> 2) * 4;            // first of 4 n-tiles

    if (wave == 0) {                              // one TDM op stages whole W tile
        tdm_load_weights(Wt, KP, wlds);
        __builtin_amdgcn_s_wait_tensorcnt((short)0);
    }
    __syncthreads();

    f32x8 acc[2][4];
#pragma unroll
    for (int m = 0; m < 2; ++m)
#pragma unroll
        for (int j = 0; j < 4; ++j)
#pragma unroll
            for (int i = 0; i < 8; ++i) acc[m][j][i] = 0.0f;

#pragma unroll
    for (int ks = 0; ks < KSTEPS; ++ks) {
        const int k0 = ks * 32;
        if (ks + 1 < KSTEPS)                      // gfx1250 global_prefetch_b8
            __builtin_prefetch(A + (size_t)(rows0 + (lane & 15)) * KP + k0 + 32, 0, 1);
        bf16x16 a0 = load_a_frag(A, KP, rows0, k0);
        bf16x16 a1 = load_a_frag(A, KP, rows0 + 16, k0);
#pragma unroll
        for (int j = 0; j < 4; ++j) {
            bf16x16 b = load_b_frag(wlds, KP, (nb + j) * 16, k0);
            acc[0][j] = wmma_bf16(a0, b, acc[0][j]);
            acc[1][j] = wmma_bf16(a1, b, acc[1][j]);
        }
    }

    const int cidx  = lane & 15;
    const int rhalf = (lane & 16) ? 8 : 0;
#pragma unroll
    for (int m = 0; m < 2; ++m) {
        const int rbase = rows0 + m * 16 + rhalf;
#pragma unroll
        for (int j = 0; j < 4; ++j) {
            const int col = (nb + j) * 16 + cidx;
            const float bb = (MODE == 2) ? bias[col] : 0.0f;
#pragma unroll
            for (int i = 0; i < 8; ++i) {
                float v = acc[m][j][i];
                size_t o = (size_t)(rbase + i) * 128 + col;
                if (MODE == 0) {
                    out1[o] = (__bf16)v;
                    out0[o] = (__bf16)fmaxf(v, 0.0f);
                } else if (MODE == 1) {
                    v += (float)addend[o];
                    out0[o] = (__bf16)fmaxf(v, 0.0f);
                } else {
                    out0[o] = (__bf16)fmaxf(v + bb, 0.0f);
                }
            }
        }
    }
}

// ---------------------------------------------------------------------------
// Attention readout: one molecule (S=32, H=128) per block, 256 threads.
// out[m][n] = (1/32) * sum_s ( h[s][n] + relu((softmax(hWa hT) h) Wb + b_b)[s][n] )
// ---------------------------------------------------------------------------
__global__ __launch_bounds__(256) void attention_readout(const __bf16* __restrict__ ah,
                                                         const __bf16* __restrict__ Wt_a,
                                                         const __bf16* __restrict__ Wt_b,
                                                         const float* __restrict__ b_b,
                                                         float* __restrict__ out) {
    __shared__ __bf16 h_s[32][128];
    __shared__ __bf16 hT_s[128][32];
    __shared__ __bf16 q_s[32][128];     // Q, later reused for att@h
    __shared__ float  sc_s[32][32];
    __shared__ __bf16 att_s[32][32];
    __shared__ float  acc_s[32][128];

    const int tid  = threadIdx.x;
    const int wave = tid >> 5;
    const int lane = tid & 31;
    const size_t molbase = (size_t)blockIdx.x * 32 * 128;

    for (int e = tid; e < 32 * 128; e += 256) {
        int s = e >> 7, c = e & 127;
        __bf16 v = ah[molbase + e];
        h_s[s][c] = v;
        hT_s[c][s] = v;
    }
    __syncthreads();

    // ---- Q = h @ W_a  (2 mtiles x 8 ntiles over 8 waves, 2 ntiles each)
    {
        const int mt = wave >> 2;
        const int nb = (wave & 3) * 2;
        f32x8 acc[2];
#pragma unroll
        for (int j = 0; j < 2; ++j)
#pragma unroll
            for (int i = 0; i < 8; ++i) acc[j][i] = 0.0f;
#pragma unroll
        for (int ks = 0; ks < 4; ++ks) {
            bf16x16 a = load_a_frag(&h_s[0][0], 128, mt * 16, ks * 32);
#pragma unroll
            for (int j = 0; j < 2; ++j) {
                bf16x16 b = load_b_frag(Wt_a, 128, (nb + j) * 16, ks * 32);
                acc[j] = wmma_bf16(a, b, acc[j]);
            }
        }
        const int rbase = mt * 16 + ((lane & 16) ? 8 : 0);
        const int cidx  = lane & 15;
#pragma unroll
        for (int j = 0; j < 2; ++j)
#pragma unroll
            for (int i = 0; i < 8; ++i)
                q_s[rbase + i][(nb + j) * 16 + cidx] = (__bf16)acc[j][i];
    }
    __syncthreads();

    // ---- scores = Q @ h^T   (B[k][n] = h[n][k] -> h_s rows directly)
    if (wave < 4) {
        const int mt = wave >> 1, nt = wave & 1;
        f32x8 acc;
#pragma unroll
        for (int i = 0; i < 8; ++i) acc[i] = 0.0f;
#pragma unroll
        for (int ks = 0; ks < 4; ++ks) {
            bf16x16 a = load_a_frag(&q_s[0][0], 128, mt * 16, ks * 32);
            bf16x16 b = load_b_frag(&h_s[0][0], 128, nt * 16, ks * 32);
            acc = wmma_bf16(a, b, acc);
        }
        const int rbase = mt * 16 + ((lane & 16) ? 8 : 0);
        const int cc = nt * 16 + (lane & 15);
#pragma unroll
        for (int i = 0; i < 8; ++i) sc_s[rbase + i][cc] = acc[i];
    }
    __syncthreads();

    // ---- softmax over rows (32 rows, one thread each)
    if (tid < 32) {
        float m = -1e30f;
#pragma unroll
        for (int c = 0; c < 32; ++c) m = fmaxf(m, sc_s[tid][c]);
        float sum = 0.0f;
        float ex[32];
#pragma unroll
        for (int c = 0; c < 32; ++c) { ex[c] = __expf(sc_s[tid][c] - m); sum += ex[c]; }
        float inv = 1.0f / sum;
#pragma unroll
        for (int c = 0; c < 32; ++c) att_s[tid][c] = (__bf16)(ex[c] * inv);
    }
    __syncthreads();

    // ---- P = att @ h  (K=32, B[k][n] = h[k][n] -> hT_s rows), store into q_s
    {
        const int mt = wave >> 2;
        const int nb = (wave & 3) * 2;
        bf16x16 a = load_a_frag(&att_s[0][0], 32, mt * 16, 0);
        f32x8 acc[2];
#pragma unroll
        for (int j = 0; j < 2; ++j) {
#pragma unroll
            for (int i = 0; i < 8; ++i) acc[j][i] = 0.0f;
            bf16x16 b = load_b_frag(&hT_s[0][0], 32, (nb + j) * 16, 0);
            acc[j] = wmma_bf16(a, b, acc[j]);
        }
        __syncthreads();
        const int rbase = mt * 16 + ((lane & 16) ? 8 : 0);
        const int cidx  = lane & 15;
#pragma unroll
        for (int j = 0; j < 2; ++j)
#pragma unroll
            for (int i = 0; i < 8; ++i)
                q_s[rbase + i][(nb + j) * 16 + cidx] = (__bf16)acc[j][i];
    }
    __syncthreads();

    // ---- att_h = relu(P @ W_b + b_b) -> acc_s (f32)
    {
        const int mt = wave >> 2;
        const int nb = (wave & 3) * 2;
        f32x8 acc[2];
#pragma unroll
        for (int j = 0; j < 2; ++j)
#pragma unroll
            for (int i = 0; i < 8; ++i) acc[j][i] = 0.0f;
#pragma unroll
        for (int ks = 0; ks < 4; ++ks) {
            bf16x16 a = load_a_frag(&q_s[0][0], 128, mt * 16, ks * 32);
#pragma unroll
            for (int j = 0; j < 2; ++j) {
                bf16x16 b = load_b_frag(Wt_b, 128, (nb + j) * 16, ks * 32);
                acc[j] = wmma_bf16(a, b, acc[j]);
            }
        }
        const int rbase = mt * 16 + ((lane & 16) ? 8 : 0);
        const int cidx  = lane & 15;
#pragma unroll
        for (int j = 0; j < 2; ++j) {
            const int col = (nb + j) * 16 + cidx;
            const float bb = b_b[col];
#pragma unroll
            for (int i = 0; i < 8; ++i)
                acc_s[rbase + i][col] = fmaxf(acc[j][i] + bb, 0.0f);
        }
    }
    __syncthreads();

    // ---- mol_vec[n] = sum_s (h + att_h) / 32
    if (tid < 128) {
        float sum = 0.0f;
#pragma unroll
        for (int s = 0; s < 32; ++s) sum += (float)h_s[s][tid] + acc_s[s][tid];
        out[(size_t)blockIdx.x * 128 + tid] = sum * (1.0f / 32.0f);
    }
}

// ---------------------------------------------------------------------------
// Host orchestration
// ---------------------------------------------------------------------------
extern "C" void kernel_launch(void* const* d_in, const int* in_sizes, int n_in,
                              void* d_out, int out_size, void* d_ws, size_t ws_size,
                              hipStream_t stream) {
    (void)in_sizes; (void)n_in; (void)out_size; (void)ws_size;

    const float* f_atoms = (const float*)d_in[0];   // [A,133]
    const float* f_bonds = (const float*)d_in[1];   // [B,147]
    const float* W_i     = (const float*)d_in[2];   // [147,128]
    const float* W_h     = (const float*)d_in[3];   // [128,128]
    const float* W_o     = (const float*)d_in[4];   // [261,128]
    const float* b_o     = (const float*)d_in[5];   // [128]
    const float* W_a     = (const float*)d_in[6];   // [128,128]
    const float* W_b     = (const float*)d_in[7];   // [128,128]
    const float* b_b     = (const float*)d_in[8];   // [128]
    const int*   a2b     = (const int*)d_in[9];     // [A,6]
    const int*   b2a     = (const int*)d_in[10];    // [B]
    const int*   b2revb  = (const int*)d_in[11];    // [B]

    constexpr int A = 262144, B = 524288;

    char* ws = (char*)d_ws;
    size_t off = 0;
    auto alloc = [&](size_t bytes) -> void* {
        void* p = ws + off;
        off += (bytes + 255) & ~(size_t)255;
        return p;
    };

    __bf16* fb   = (__bf16*)alloc((size_t)B * 160 * 2);  // bonds bf16 padded; reused: bmsg, ainput
    __bf16* inp  = (__bf16*)alloc((size_t)B * 128 * 2);  // inputs (pre-relu), bf16
    __bf16* msg0 = (__bf16*)alloc((size_t)B * 128 * 2);  // message ping; reused: atom_hiddens
    __bf16* msg1 = (__bf16*)alloc((size_t)B * 128 * 2);  // message pong
    __bf16* amsg = (__bf16*)alloc((size_t)A * 128 * 2);  // atom message
    __bf16* wt_i = (__bf16*)alloc((size_t)128 * 160 * 2);
    __bf16* wt_h = (__bf16*)alloc((size_t)128 * 128 * 2);
    __bf16* wt_o = (__bf16*)alloc((size_t)128 * 288 * 2);
    __bf16* wt_a = (__bf16*)alloc((size_t)128 * 128 * 2);
    __bf16* wt_b = (__bf16*)alloc((size_t)128 * 128 * 2);
    __bf16* bmsg   = fb;      // alive only between GEMM1 and last W_h GEMM
    __bf16* ainput = fb;      // alive after last bmsg use; 151MB <= 168MB region
    __bf16* ah     = msg0;    // msg0 dead after final message lands in msg1

    // 1) pack transposed/padded bf16 weights
    pack_wT<<<(128 * 160) / 256, 256, 0, stream>>>(W_i, 147, 160, wt_i);
    pack_wT<<<(128 * 128) / 256, 256, 0, stream>>>(W_h, 128, 128, wt_h);
    pack_wT<<<(128 * 288) / 256, 256, 0, stream>>>(W_o, 261, 288, wt_o);
    pack_wT<<<(128 * 128) / 256, 256, 0, stream>>>(W_a, 128, 128, wt_a);
    pack_wT<<<(128 * 128) / 256, 256, 0, stream>>>(W_b, 128, 128, wt_b);

    // 2) f_bonds f32[147] -> bf16[160] padded
    pack_pad_bf16<<<(unsigned)((size_t)B * 160 / 256), 256, 0, stream>>>(f_bonds, 147, 160, fb);

    // 3) inputs = f_bonds @ W_i ; message = relu(inputs)
    gemm_bf16<5, 0, 160><<<B / 128, 256, 160 * 128 * 2, stream>>>(fb, wt_i, nullptr, nullptr, msg0, inp);

    // 4) 3 message-passing iterations
    __bf16* cur = msg0;
    __bf16* nxt = msg1;
    for (int t = 0; t < 3; ++t) {
        gather_amsg<<<(A * 32) / 256, 256, 0, stream>>>(cur, a2b, amsg);
        bond_msg<<<(B * 32) / 256, 256, 0, stream>>>(amsg, cur, b2a, b2revb, bmsg);
        gemm_bf16<4, 1, 128><<<B / 128, 256, 128 * 128 * 2, stream>>>(bmsg, wt_h, inp, nullptr, nxt, nullptr);
        __bf16* tmp = cur; cur = nxt; nxt = tmp;
    }

    // 5) final atom aggregation + concat + W_o GEMM
    gather_amsg<<<(A * 32) / 256, 256, 0, stream>>>(cur, a2b, amsg);
    pack_ainput<<<(unsigned)((size_t)A * 288 / 256), 256, 0, stream>>>(f_atoms, amsg, ainput);
    gemm_bf16<9, 2, 288><<<A / 128, 256, 288 * 128 * 2, stream>>>(ainput, wt_o, nullptr, b_o, ah, nullptr);

    // 6) per-molecule self-attention readout -> d_out [8192,128] f32
    attention_readout<<<A / 32, 256, 0, stream>>>(ah, wt_a, wt_b, b_b, (float*)d_out);
}
